// MHA_35407710388493
// MI455X (gfx1250) — matrix operand order repaired and checked by reading, not compiled, not run on hardware
//
#include <hip/hip_runtime.h>
#include <hip/hip_bf16.h>

#define SEQ     4096
#define DIM     1024
#define NHEADS  16
#define NKV     4
#define HDIM    64

typedef __attribute__((ext_vector_type(16))) __bf16 v16bf;
typedef __attribute__((ext_vector_type(8)))  __bf16 v8bf;
typedef __attribute__((ext_vector_type(8)))  float  v8f;
typedef __attribute__((ext_vector_type(4)))  float  f32x4;
typedef __attribute__((ext_vector_type(4)))  unsigned int u32x4;
typedef __attribute__((ext_vector_type(8)))  int    i32x8;
typedef __attribute__((ext_vector_type(4)))  int    i32x4;

// ---------- bf16 <-> f32 via native converts (v_cvt_pk_bf16_f32) ----------
static __device__ __forceinline__ __bf16 f2bf(float f)  { return (__bf16)f; }
static __device__ __forceinline__ float  bf2f(__bf16 h) { return (float)h; }

// ---------- DPP-based 16-lane xor reductions (no LDS/bpermute traffic) ----------
#define DPP8_SEL(l0,l1,l2,l3,l4,l5,l6,l7) \
    ((l0)|((l1)<<3)|((l2)<<6)|((l3)<<9)|((l4)<<12)|((l5)<<15)|((l6)<<18)|((l7)<<21))

static __device__ __forceinline__ float dpp_xor1(float x) {
    return __builtin_bit_cast(float, __builtin_amdgcn_mov_dpp8(
        __builtin_bit_cast(int, x), DPP8_SEL(1,0,3,2,5,4,7,6)));
}
static __device__ __forceinline__ float dpp_xor2(float x) {
    return __builtin_bit_cast(float, __builtin_amdgcn_mov_dpp8(
        __builtin_bit_cast(int, x), DPP8_SEL(2,3,0,1,6,7,4,5)));
}
static __device__ __forceinline__ float dpp_xor4(float x) {
    return __builtin_bit_cast(float, __builtin_amdgcn_mov_dpp8(
        __builtin_bit_cast(int, x), DPP8_SEL(4,5,6,7,0,1,2,3)));
}
static __device__ __forceinline__ float dpp_xor8(float x) {
    // ROW_ROR:8 within a 16-lane row == lane ^ 8
    return __builtin_bit_cast(float, __builtin_amdgcn_update_dpp(
        0, __builtin_bit_cast(int, x), 0x128, 0xf, 0xf, true));
}
static __device__ __forceinline__ float red_max16(float x) {
    x = fmaxf(x, dpp_xor1(x));
    x = fmaxf(x, dpp_xor2(x));
    x = fmaxf(x, dpp_xor4(x));
    x = fmaxf(x, dpp_xor8(x));
    return x;
}
static __device__ __forceinline__ float red_sum16(float x) {
    x += dpp_xor1(x);
    x += dpp_xor2(x);
    x += dpp_xor4(x);
    x += dpp_xor8(x);
    return x;
}

// ---------- WMMA fragment loaders: every load is 2x 128-bit ----------
// A 16x32 (row-major, stride elems): lane m=lane%16, k in {kb..kb+7, kb+16..kb+23}, kb=8*(lane/16)
static __device__ __forceinline__ v16bf ld_fragA(const __bf16* p, int stride) {
    int lane = threadIdx.x & 31;
    const __bf16* r = p + (lane & 15) * stride + ((lane >> 4) << 3);
    v8bf lo = *(const v8bf*)(r);
    v8bf hi = *(const v8bf*)(r + 16);
    return __builtin_shufflevector(lo, hi, 0,1,2,3,4,5,6,7,8,9,10,11,12,13,14,15);
}
// B 32x16 from n-major tile (k contiguous within an n-row of `stride` elems):
// lane n=lane%16, k in {kb..kb+7, kb+16..kb+23}, kb=8*(lane/16)
static __device__ __forceinline__ v16bf ld_fragB(const __bf16* p, int stride) {
    int lane = threadIdx.x & 31;
    const __bf16* r = p + (lane & 15) * stride + ((lane >> 4) << 3);
    v8bf lo = *(const v8bf*)(r);
    v8bf hi = *(const v8bf*)(r + 16);
    return __builtin_shufflevector(lo, hi, 0,1,2,3,4,5,6,7,8,9,10,11,12,13,14,15);
}

static __device__ __forceinline__ v8f wmma_bf16(v16bf a, v16bf b, v8f c) {
    return __builtin_amdgcn_wmma_f32_16x16x32_bf16(false, a, false, b, (short)0, c, false, false);
}

// ---------- Tensor Data Mover: 2-D tile load (bf16) global -> LDS ----------
// D# per CDNA5 ISA 8.3/8.4: group0 {count=1, lds_addr, global_addr, type=2},
// group1 {data_size=1(2B), tensor_dim0/1, tile_dim0/1, tensor_dim0_stride}.
static __device__ __forceinline__ void tdm_load_2d_bf16(unsigned lds_off, const void* gptr,
                                                        unsigned tile_w, unsigned tile_h,
                                                        unsigned row_stride, unsigned tensor_h) {
    unsigned long long ga = (unsigned long long)gptr;
    u32x4 g0;
    g0[0] = 1u;                                               // count=1, is_restore=0
    g0[1] = lds_off;                                          // lds_addr
    g0[2] = (unsigned)(ga & 0xFFFFFFFFu);                     // global_addr[31:0]
    g0[3] = (unsigned)((ga >> 32) & 0x01FFFFFFu) | (2u << 30);// global_addr[56:32], type=2
    i32x8 g1;
    g1[0] = (int)(1u << 16);                                  // workgroup_mask=0, data_size=2B
    g1[1] = (int)((row_stride & 0xFFFFu) << 16);              // tensor_dim0[15:0]
    g1[2] = (int)(((row_stride >> 16) & 0xFFFFu) | ((tensor_h & 0xFFFFu) << 16));
    g1[3] = (int)(((tensor_h >> 16) & 0xFFFFu) | ((tile_w & 0xFFFFu) << 16)); // tile_dim0
    g1[4] = (int)(tile_h & 0xFFFFu);                          // tile_dim1, tile_dim2=0
    g1[5] = (int)row_stride;                                  // tensor_dim0_stride[31:0]
    g1[6] = 0;
    g1[7] = 0;
    i32x4 z4 = {0, 0, 0, 0};
#if defined(__clang_major__) && (__clang_major__ >= 23)
    i32x8 z8 = {0, 0, 0, 0, 0, 0, 0, 0};
    __builtin_amdgcn_tensor_load_to_lds(g0, g1, z4, z4, z8, 0);
#else
    __builtin_amdgcn_tensor_load_to_lds(g0, g1, z4, z4, 0);
#endif
}

// =====================================================================
// Kernel 1: C[4096 x N] = A(f32, 4096x1024) * W(f32, 1024xN) -> bf16
// block = 256 threads (8 waves), tile 256x64, K chunk 32.
// Each wave owns 32 rows (2 A-frags) so every B-frag feeds 2 WMMAs.
// N is a template parameter so strided W loads use immediate offsets.
// =====================================================================
template <int N>
__global__ __launch_bounds__(256) void qkv_gemm(const float* __restrict__ A,
                                                const float* __restrict__ W,
                                                __bf16* __restrict__ out) {
    __shared__ __attribute__((aligned(16))) __bf16 As[256 * 32];  // row-major, A operand
    __shared__ __attribute__((aligned(16))) __bf16 Wt[64 * 32];   // n-major (transposed), B operand
    int tid  = threadIdx.x;
    int w    = tid >> 5, lane = tid & 31;
    int half = lane >> 4, n = lane & 15;
    int rowbase = blockIdx.y * 256;
    int colbase = blockIdx.x * 64;

    v8f acc0[4] = {v8f{}, v8f{}, v8f{}, v8f{}};
    v8f acc1[4] = {v8f{}, v8f{}, v8f{}, v8f{}};

    for (int k0 = 0; k0 < DIM; k0 += 32) {
        // ---- stage A: 256x32 f32 -> bf16, 16B vector ld/st, 4 chunks/thread ----
#pragma unroll
        for (int i = 0; i < 4; ++i) {
            int chunk = tid + (i << 8);              // 0..1023
            int r = chunk >> 2, kc = chunk & 3;
            const float* src = A + (size_t)(rowbase + r) * DIM + k0 + kc * 8;
            f32x4 lo = *(const f32x4*)(src);
            f32x4 hi = *(const f32x4*)(src + 4);
            v8bf d;
#pragma unroll
            for (int j = 0; j < 4; ++j) { d[j] = f2bf(lo[j]); d[4 + j] = f2bf(hi[j]); }
            *(v8bf*)(As + r * 32 + kc * 8) = d;
        }
        // ---- stage W transposed: Wt[c][k]; constant stride N -> clause'd loads ----
        {
            int c = tid & 63, kc = tid >> 6;
            const float* src = W + (size_t)(k0 + kc * 8) * N + colbase + c;
            v8bf d;
#pragma unroll
            for (int j = 0; j < 8; ++j)
                d[j] = f2bf(src[(size_t)j * N]);
            *(v8bf*)(Wt + c * 32 + kc * 8) = d;
        }
        if (k0 + 32 < DIM)                           // gfx1250 global_prefetch_b8
            __builtin_prefetch(&A[(size_t)(rowbase + tid) * DIM + k0 + 32], 0, 0);
        __syncthreads();

        v16bf aLo = ld_fragA(As + (w * 32 + 0)  * 32, 32);
        v16bf aHi = ld_fragA(As + (w * 32 + 16) * 32, 32);
#pragma unroll
        for (int t = 0; t < 4; ++t) {
            v16bf b = ld_fragB(Wt + t * 16 * 32, 32);
            acc0[t] = wmma_bf16(aLo, b, acc0[t]);
            acc1[t] = wmma_bf16(aHi, b, acc1[t]);
        }
        __syncthreads();
    }
#pragma unroll
    for (int t = 0; t < 4; ++t)
#pragma unroll
        for (int r = 0; r < 8; ++r) {
            int col = colbase + t * 16 + n;
            out[(size_t)(rowbase + w * 32 + 0  + r + half * 8) * N + col] = f2bf(acc0[t][r]);
            out[(size_t)(rowbase + w * 32 + 16 + r + half * 8) * N + col] = f2bf(acc1[t][r]);
        }
}

// =====================================================================
// Kernel 2: RoPE in place on bf16 [SEQ][H*64]
// =====================================================================
__global__ __launch_bounds__(256) void rope_kernel(__bf16* __restrict__ buf, int H) {
    int idx = blockIdx.x * blockDim.x + threadIdx.x;     // over SEQ*H*32 pairs
    int total = SEQ * H * 32;
    if (idx >= total) return;
    int i = idx & 31;
    int h = (idx >> 5) % H;
    int s = idx / (32 * H);
    float theta = powf(1000.0f, -(2.0f * (float)i) / 64.0f);
    float ang = (float)s * theta;
    float c = cosf(ang), sn = sinf(ang);
    __bf16* p = buf + (size_t)s * (H * 64) + h * 64 + 2 * i;
    float x1 = bf2f(p[0]), x2 = bf2f(p[1]);
    p[0] = f2bf(x1 * c - x2 * sn);
    p[1] = f2bf(x1 * sn + x2 * c);
}

// =====================================================================
// Kernel 3: causal flash attention, one (head, 128-query tile) per block
// q: bf16 [SEQ][1024] (head h at cols h*64), k/v: bf16 [SEQ][256]
// K tile staged by the Tensor Data Mover; V tile staged transposed.
// =====================================================================
__global__ __launch_bounds__(256) void flash_attn(const __bf16* __restrict__ q,
                                                  const __bf16* __restrict__ k,
                                                  const __bf16* __restrict__ v,
                                                  __bf16* __restrict__ attn_out) {
    __shared__ __attribute__((aligned(16))) __bf16 Ks[32 * 64];   // [key][d]  (n-major for QK^T)
    __shared__ __attribute__((aligned(16))) __bf16 Vt[64 * 32];   // [d][key]  (n-major for P*V)
    __shared__ __attribute__((aligned(16))) __bf16 Ps[8][16 * 32];// per-wave P (A operand)

    int h  = blockIdx.x;
    int qt = blockIdx.y;
    int tid = threadIdx.x;
    int w = tid >> 5, lane = tid & 31;
    int half = lane >> 4, n = lane & 15;
    int hk = h >> 2;                       // GQA: 4 query heads per kv head
    int qbase = qt * 128 + w * 16;
    unsigned ldsKs = (unsigned)(size_t)(&Ks[0]);

    // Q fragments for this wave's 16 rows: d = 0..31 and d = 32..63
    const __bf16* qp = q + (size_t)qbase * (NHEADS * HDIM) + h * HDIM;
    v16bf a0 = ld_fragA(qp, NHEADS * HDIM);
    v16bf a1 = ld_fragA(qp + 32, NHEADS * HDIM);

    v8f o0 = {}, o1 = {}, o2 = {}, o3 = {};
    float mrow[8], lrow[8];
#pragma unroll
    for (int r = 0; r < 8; ++r) { mrow[r] = -1e30f; lrow[r] = 0.0f; }

    const float scale = 0.125f;            // HEAD_DIM^-0.5
    int kend = qt * 128 + 128;             // causal: keys <= max query row in block

    for (int kb = 0; kb < kend; kb += 32) {
        // ---- K tile 32x64 via TDM (wave 0 issues; EXEC-independent DMA) ----
        if (w == 0) {
            const __bf16* kptr = k + (size_t)kb * (NKV * HDIM) + hk * HDIM;
            tdm_load_2d_bf16(ldsKs, kptr, /*tile_w=*/HDIM, /*tile_h=*/32,
                             /*row_stride=*/NKV * HDIM, /*tensor_h=*/SEQ);
        }
        // ---- V tile transposed: Vt[d][key], one b128 store per thread ----
        {
            int d = tid & 63, kc = tid >> 6;
            const __bf16* src = v + (size_t)(kb + kc * 8) * (NKV * HDIM) + hk * HDIM + d;
            v8bf t;
#pragma unroll
            for (int j = 0; j < 8; ++j)
                t[j] = src[j * (NKV * HDIM)];
            *(v8bf*)(Vt + d * 32 + kc * 8) = t;
        }
        if (w == 0) __builtin_amdgcn_s_wait_tensorcnt(0);
        __syncthreads();

        // ---- S = Q * K^T : two 16x16 subtiles, K-dim = head dim (2 x 32-chunk) ----
        v8f s0 = {}, s1 = {};
        s0 = wmma_bf16(a0, ld_fragB(Ks + 0 * 16 * 64 + 0,  64), s0);
        s0 = wmma_bf16(a1, ld_fragB(Ks + 0 * 16 * 64 + 32, 64), s0);
        s1 = wmma_bf16(a0, ld_fragB(Ks + 1 * 16 * 64 + 0,  64), s1);
        s1 = wmma_bf16(a1, ld_fragB(Ks + 1 * 16 * 64 + 32, 64), s1);

        // ---- online softmax (rows span 16 lanes of this half-wave, DPP reductions) ----
#pragma unroll
        for (int r = 0; r < 8; ++r) {
            int row = qbase + r + half * 8;
            float v0 = s0[r] * scale;
            float v1 = s1[r] * scale;
            if (kb + n      > row) v0 = -1e30f;
            if (kb + 16 + n > row) v1 = -1e30f;
            float mx   = red_max16(fmaxf(v0, v1));
            float mnew = fmaxf(mrow[r], mx);
            float fac  = __expf(mrow[r] - mnew);
            float p0   = __expf(v0 - mnew);
            float p1   = __expf(v1 - mnew);
            float rs   = red_sum16(p0 + p1);
            lrow[r] = lrow[r] * fac + rs;
            mrow[r] = mnew;
            o0[r] *= fac; o1[r] *= fac; o2[r] *= fac; o3[r] *= fac;
            Ps[w][(r + half * 8) * 32 + n]      = f2bf(p0);
            Ps[w][(r + half * 8) * 32 + 16 + n] = f2bf(p1);
        }

        // ---- O += P * V  (K-dim = 32 keys) ----
        v16bf pa = ld_fragA(&Ps[w][0], 32);
        o0 = wmma_bf16(pa, ld_fragB(Vt + 0 * 16 * 32, 32), o0);
        o1 = wmma_bf16(pa, ld_fragB(Vt + 1 * 16 * 32, 32), o1);
        o2 = wmma_bf16(pa, ld_fragB(Vt + 2 * 16 * 32, 32), o2);
        o3 = wmma_bf16(pa, ld_fragB(Vt + 3 * 16 * 32, 32), o3);
        __syncthreads();
    }

    // ---- normalize and write attn_out [SEQ][1024] ----
    __bf16* op = attn_out + (size_t)qbase * (NHEADS * HDIM) + h * HDIM;
#pragma unroll
    for (int r = 0; r < 8; ++r) {
        float inv = 1.0f / lrow[r];
        size_t ro = (size_t)(r + half * 8) * (NHEADS * HDIM);
        op[ro + 0 * 16 + n] = f2bf(o0[r] * inv);
        op[ro + 1 * 16 + n] = f2bf(o1[r] * inv);
        op[ro + 2 * 16 + n] = f2bf(o2[r] * inv);
        op[ro + 3 * 16 + n] = f2bf(o3[r] * inv);
    }
}

// =====================================================================
// Kernel 4: out[4096x1024] f32 = attn(bf16, 4096x1024) * Wo(f32, 1024x1024)
// tile 256x64, 2 A-frags per wave (B-frag reuse), constant strides.
// =====================================================================
__global__ __launch_bounds__(256) void out_gemm(const __bf16* __restrict__ A,
                                                const float* __restrict__ W,
                                                float* __restrict__ out) {
    __shared__ __attribute__((aligned(16))) __bf16 As[256 * 32];
    __shared__ __attribute__((aligned(16))) __bf16 Wt[64 * 32];
    int tid  = threadIdx.x;
    int w    = tid >> 5, lane = tid & 31;
    int half = lane >> 4, n = lane & 15;
    int rowbase = blockIdx.y * 256;
    int colbase = blockIdx.x * 64;

    v8f acc0[4] = {v8f{}, v8f{}, v8f{}, v8f{}};
    v8f acc1[4] = {v8f{}, v8f{}, v8f{}, v8f{}};

    for (int k0 = 0; k0 < DIM; k0 += 32) {
        // ---- stage A (bf16): pure 16B copies, 4 chunks/thread ----
#pragma unroll
        for (int i = 0; i < 4; ++i) {
            int chunk = tid + (i << 8);
            int r = chunk >> 2, kc = chunk & 3;
            *(uint4*)(As + r * 32 + kc * 8) =
                *(const uint4*)(A + (size_t)(rowbase + r) * DIM + k0 + kc * 8);
        }
        // ---- stage Wo transposed (constant stride DIM -> clause'd loads) ----
        {
            int c = tid & 63, kc = tid >> 6;
            const float* src = W + (size_t)(k0 + kc * 8) * DIM + colbase + c;
            v8bf d;
#pragma unroll
            for (int j = 0; j < 8; ++j)
                d[j] = f2bf(src[(size_t)j * DIM]);
            *(v8bf*)(Wt + c * 32 + kc * 8) = d;
        }
        if (k0 + 32 < DIM)
            __builtin_prefetch(&W[(size_t)(k0 + 32 + (tid >> 6)) * DIM + colbase + (tid & 63)], 0, 0);
        __syncthreads();

        v16bf aLo = ld_fragA(As + (w * 32 + 0)  * 32, 32);
        v16bf aHi = ld_fragA(As + (w * 32 + 16) * 32, 32);
#pragma unroll
        for (int t = 0; t < 4; ++t) {
            v16bf b = ld_fragB(Wt + t * 16 * 32, 32);
            acc0[t] = wmma_bf16(aLo, b, acc0[t]);
            acc1[t] = wmma_bf16(aHi, b, acc1[t]);
        }
        __syncthreads();
    }
#pragma unroll
    for (int t = 0; t < 4; ++t)
#pragma unroll
        for (int r = 0; r < 8; ++r) {
            int col = colbase + t * 16 + n;
            out[(size_t)(rowbase + w * 32 + 0  + r + half * 8) * DIM + col] = acc0[t][r];
            out[(size_t)(rowbase + w * 32 + 16 + r + half * 8) * DIM + col] = acc1[t][r];
        }
}

// =====================================================================
extern "C" void kernel_launch(void* const* d_in, const int* in_sizes, int n_in,
                              void* d_out, int out_size, void* d_ws, size_t ws_size,
                              hipStream_t stream) {
    const float* x  = (const float*)d_in[0];
    // d_in[1] = mask (unused: causal mask computed analytically)
    const float* Wq = (const float*)d_in[2];
    const float* Wk = (const float*)d_in[3];
    const float* Wv = (const float*)d_in[4];
    const float* Wo = (const float*)d_in[5];
    float* out = (float*)d_out;

    char* ws = (char*)d_ws;
    __bf16* qb = (__bf16*)(ws);                               // 4096*1024*2 = 8 MiB
    __bf16* kb = (__bf16*)(ws + (8ull  << 20));               // 4096* 256*2 = 2 MiB
    __bf16* vb = (__bf16*)(ws + (10ull << 20));               // 4096* 256*2 = 2 MiB
    __bf16* ab = (__bf16*)(ws + (12ull << 20));               // 4096*1024*2 = 8 MiB

    dim3 blk(256);
    // QKV projections (f32 -> bf16 with f32 WMMA accumulation)
    qkv_gemm<NHEADS * HDIM><<<dim3(16, 16), blk, 0, stream>>>(x, Wq, qb);
    qkv_gemm<NKV * HDIM>   <<<dim3(4,  16), blk, 0, stream>>>(x, Wk, kb);
    qkv_gemm<NKV * HDIM>   <<<dim3(4,  16), blk, 0, stream>>>(x, Wv, vb);
    // RoPE on Q and K
    rope_kernel<<<(SEQ * NHEADS * 32 + 255) / 256, blk, 0, stream>>>(qb, NHEADS);
    rope_kernel<<<(SEQ * NKV    * 32 + 255) / 256, blk, 0, stream>>>(kb, NKV);
    // Causal flash attention (per head, per 128-query tile)
    flash_attn<<<dim3(NHEADS, SEQ / 128), blk, 0, stream>>>(qb, kb, vb, ab);
    // Output projection -> f32
    out_gemm<<<dim3(16, 16), blk, 0, stream>>>(ab, Wo, out);
}
